// DiagWinAttention_7112465842439
// MI455X (gfx1250) — matrix operand order repaired and checked by reading, not compiled, not run on hardware
//
#include <hip/hip_runtime.h>

typedef _Float16 v16h __attribute__((ext_vector_type(16)));
typedef _Float16 v8h  __attribute__((ext_vector_type(8)));
typedef _Float16 v4h  __attribute__((ext_vector_type(4)));
typedef float    v8f  __attribute__((ext_vector_type(8)));

#define NTOK 64
#define EMB  96
#define NHEAD 6
#define CHD  16
#define QK_SCALE 0.25f     // 16^-0.5
#define LN_EPS 1e-5f

// LDS carve (bytes). Phase B peak: Q+K+Vt (f16) + mask(f32) + attn(f16,6 heads) + out(f32)
#define OFF_Q    0          // 64*96*2  = 12288   (phase A,B; aliased by X-hat in C/D)
#define OFF_K    12288      // 64*96*2  = 12288   (phase A,B; aliased by W-f16 in C/D)
#define OFF_VT   24576      // 96*64*2  = 12288   (phase A,B; W-f16 spills into here)
#define OFF_MASK 36864      // 64*64*4  = 16384   (phase B)
#define OFF_ATTN 53248      // 6*64*64*2= 49152   (phase B, per-head private)
#define OFF_OUT  102400     // 64*96*4  = 24576   (phase B-end .. C)
#define SMEM_BYTES 126976

__global__ __launch_bounds__(192, 1)
void winattn_fused(const float* __restrict__ gq,   const float* __restrict__ gk,
                   const float* __restrict__ gv,   const float* __restrict__ gmask,
                   const float* __restrict__ gbias,const float* __restrict__ ggamma,
                   const float* __restrict__ gbeta,const float* __restrict__ gpw,
                   const float* __restrict__ gpb,  float* __restrict__ gout)
{
    __shared__ __align__(16) unsigned char smem[SMEM_BYTES];
    _Float16* sQ    = (_Float16*)(smem + OFF_Q);
    _Float16* sK    = (_Float16*)(smem + OFF_K);
    _Float16* sVt   = (_Float16*)(smem + OFF_VT);
    float*    sMask = (float*)   (smem + OFF_MASK);
    _Float16* sAttn = (_Float16*)(smem + OFF_ATTN);
    float*    sOut  = (float*)   (smem + OFF_OUT);

    const int tid   = threadIdx.x;
    const int lane  = tid & 31;
    const int wv    = tid >> 5;        // 0..5 (one wave per head)
    const int l16   = lane & 15;
    const int lh    = lane >> 4;       // lane half: 0 or 1
    const int rbase = lh << 3;         // D-layout row base for this half
    const int win   = blockIdx.x;

    // ---------------- Phase A: stage Q(scaled f16), K(f16), V^T(f16), mask ----
    {
        const float4* q4 = (const float4*)(gq + (size_t)win * (NTOK*EMB));
        const float4* k4 = (const float4*)(gk + (size_t)win * (NTOK*EMB));
        const float4* v4 = (const float4*)(gv + (size_t)win * (NTOK*EMB));
        v4h* dq = (v4h*)sQ;  v4h* dk = (v4h*)sK;
        for (int f = tid; f < (NTOK*EMB/4); f += 192) {
            float4 aq = q4[f];
            v4h hq; hq[0]=(_Float16)(aq.x*QK_SCALE); hq[1]=(_Float16)(aq.y*QK_SCALE);
                    hq[2]=(_Float16)(aq.z*QK_SCALE); hq[3]=(_Float16)(aq.w*QK_SCALE);
            dq[f] = hq;
            float4 ak = k4[f];
            v4h hk; hk[0]=(_Float16)ak.x; hk[1]=(_Float16)ak.y;
                    hk[2]=(_Float16)ak.z; hk[3]=(_Float16)ak.w;
            dk[f] = hk;
            float4 av = v4[f];
            int g = f*4, t = g/EMB, e = g - t*EMB;   // 4 elems never cross a row (96%4==0)
            sVt[(e+0)*NTOK + t] = (_Float16)av.x;
            sVt[(e+1)*NTOK + t] = (_Float16)av.y;
            sVt[(e+2)*NTOK + t] = (_Float16)av.z;
            sVt[(e+3)*NTOK + t] = (_Float16)av.w;
        }
        const float4* m4 = (const float4*)(gmask + (size_t)win * (NTOK*NTOK));
        float4* dm = (float4*)sMask;
        for (int f = tid; f < (NTOK*NTOK/4); f += 192) dm[f] = m4[f];
    }
    __syncthreads();

    // ---------------- Phase B: fused attention, one wave per head -------------
    {
        const int h = wv;
        const int hOff = h * CHD;
        _Float16* aRow = sAttn + h * (NTOK*NTOK);

        for (int ti = 0; ti < 4; ++ti) {
            // A operand: q-hat rows (16x16 padded to 16x32)
            v16h afrag;
            {
                int m = ti*16 + l16;
                v8h lo = *(const v8h*)(sQ + m*EMB + hOff + lh*8);
                #pragma unroll
                for (int e = 0; e < 8; ++e) { afrag[e] = lo[e]; afrag[8+e] = (_Float16)0.0f; }
            }
            v8f acc[4];
            #pragma unroll
            for (int tj = 0; tj < 4; ++tj) {
                // B operand: K^T (channels 8h..8h+7 in elements 0-7 of each lane half; elements 8-15 zero)
                v16h bfrag;
                {
                    int n = tj*16 + l16;
                    v8h b0 = *(const v8h*)(sK + n*EMB + hOff + lh*8);
                    #pragma unroll
                    for (int e = 0; e < 8; ++e) { bfrag[e] = b0[e]; bfrag[8+e] = (_Float16)0.0f; }
                }
                v8f z;
                #pragma unroll
                for (int r = 0; r < 8; ++r) z[r] = 0.0f;
                acc[tj] = __builtin_amdgcn_wmma_f32_16x16x32_f16(false, afrag, false, bfrag,
                                                                 (short)0, z, false, false);
            }
            // + relative-position bias (closed form) + per-window mask
            #pragma unroll
            for (int tj = 0; tj < 4; ++tj) {
                int n = tj*16 + l16;
                int yj = n >> 3, xj = n & 7;
                #pragma unroll
                for (int r = 0; r < 8; ++r) {
                    int m = ti*16 + rbase + r;
                    int yi = m >> 3, xi = m & 7;
                    int ridx = (yi - yj + 7)*15 + (xi - xj + 7);
                    acc[tj][r] += sMask[m*NTOK + n] + gbias[ridx*NHEAD + h];
                }
            }
            // softmax across the 64-wide row: 4 tiles folded, then 16-lane reduce
            #pragma unroll
            for (int r = 0; r < 8; ++r) {
                float x = fmaxf(fmaxf(acc[0][r], acc[1][r]), fmaxf(acc[2][r], acc[3][r]));
                for (int s = 1; s < 16; s <<= 1) x = fmaxf(x, __shfl_xor(x, s, 32));
                #pragma unroll
                for (int tj = 0; tj < 4; ++tj) acc[tj][r] = __expf(acc[tj][r] - x);
                float sum = acc[0][r] + acc[1][r] + acc[2][r] + acc[3][r];
                for (int s = 1; s < 16; s <<= 1) sum += __shfl_xor(sum, s, 32);
                float rinv = 1.0f / sum;
                #pragma unroll
                for (int tj = 0; tj < 4; ++tj) acc[tj][r] *= rinv;
            }
            // spill attn tile rows to LDS (f16, row-major) for WMMA-A reload
            #pragma unroll
            for (int tj = 0; tj < 4; ++tj)
                #pragma unroll
                for (int r = 0; r < 8; ++r) {
                    int m = ti*16 + rbase + r;
                    int n = tj*16 + l16;
                    aRow[m*NTOK + n] = (_Float16)acc[tj][r];
                }

            // PV: (16x64)@(64x16), two chained K=32 WMMAs; then +q-hat residual
            v8f oacc;
            #pragma unroll
            for (int r = 0; r < 8; ++r) oacc[r] = 0.0f;
            #pragma unroll
            for (int kb = 0; kb < 2; ++kb) {
                v16h pa;
                {
                    int m = ti*16 + l16;
                    v8h lo = *(const v8h*)(aRow + m*NTOK + kb*32 + rbase);
                    v8h hi = *(const v8h*)(aRow + m*NTOK + kb*32 + 16 + rbase);
                    #pragma unroll
                    for (int e = 0; e < 8; ++e) { pa[e] = lo[e]; pa[8+e] = hi[e]; }
                }
                v16h pbf;
                {
                    const v8h* vp = (const v8h*)(sVt + (hOff + l16)*NTOK + kb*32 + lh*8);
                    v8h b0 = vp[0], b1 = vp[2];
                    #pragma unroll
                    for (int e = 0; e < 8; ++e) { pbf[e] = b0[e]; pbf[8+e] = b1[e]; }
                }
                oacc = __builtin_amdgcn_wmma_f32_16x16x32_f16(false, pa, false, pbf,
                                                              (short)0, oacc, false, false);
            }
            #pragma unroll
            for (int r = 0; r < 8; ++r) {
                int m = ti*16 + rbase + r;
                sOut[m*EMB + hOff + l16] = oacc[r] + (float)sQ[m*EMB + hOff + l16];
            }
        }
    }
    __syncthreads();

    // ---------------- Phase C: LayerNorm + stage W as f16 (aliases dead LDS) --
    _Float16* sX = (_Float16*)(smem + OFF_Q);   // 64*96 f16, aliases sQ (dead)
    _Float16* sW = (_Float16*)(smem + OFF_K);   // 96*96 f16, aliases sK+sVt (dead)
    if (tid < NTOK) {
        const float* row = sOut + tid*EMB;
        float mu = 0.0f;
        for (int e = 0; e < EMB; ++e) mu += row[e];
        mu *= (1.0f/EMB);
        float var = 0.0f;
        for (int e = 0; e < EMB; ++e) { float d = row[e] - mu; var += d*d; }
        var *= (1.0f/EMB);
        float rs = rsqrtf(var + LN_EPS);
        for (int e = 0; e < EMB; ++e)
            sX[tid*EMB + e] = (_Float16)((row[e] - mu)*rs*ggamma[e] + gbeta[e]);
    }
    {
        const float4* w4 = (const float4*)gpw;
        v4h* dw = (v4h*)sW;
        for (int f = tid; f < (EMB*EMB/4); f += 192) {
            float4 aw = w4[f];
            v4h hw; hw[0]=(_Float16)aw.x; hw[1]=(_Float16)aw.y;
                    hw[2]=(_Float16)aw.z; hw[3]=(_Float16)aw.w;
            dw[f] = hw;
        }
    }
    __syncthreads();

    // ---------------- Phase D: projection  out = X @ W^T + b ------------------
    {
        const int o0 = wv * CHD;                 // 16 output cols per wave
        float* go = gout + (size_t)win * (NTOK*EMB);
        const float pbv = gpb[o0 + l16];
        for (int ti = 0; ti < 4; ++ti) {
            v8f acc;
            #pragma unroll
            for (int r = 0; r < 8; ++r) acc[r] = 0.0f;
            #pragma unroll
            for (int kb = 0; kb < 3; ++kb) {     // contraction E=96 = 3 x 32
                v16h xa;
                {
                    int m = ti*16 + l16;
                    v8h lo = *(const v8h*)(sX + m*EMB + kb*32 + rbase);
                    v8h hi = *(const v8h*)(sX + m*EMB + kb*32 + 16 + rbase);
                    #pragma unroll
                    for (int e = 0; e < 8; ++e) { xa[e] = lo[e]; xa[8+e] = hi[e]; }
                }
                v16h wb;
                {
                    const v8h* wp = (const v8h*)(sW + (o0 + l16)*EMB + kb*32 + lh*8);
                    v8h b0 = wp[0], b1 = wp[2];
                    #pragma unroll
                    for (int e = 0; e < 8; ++e) { wb[e] = b0[e]; wb[8+e] = b1[e]; }
                }
                acc = __builtin_amdgcn_wmma_f32_16x16x32_f16(false, xa, false, wb,
                                                             (short)0, acc, false, false);
            }
            #pragma unroll
            for (int r = 0; r < 8; ++r) {
                int m = ti*16 + rbase + r;
                go[m*EMB + o0 + l16] = acc[r] + pbv;
            }
        }
    }
}

extern "C" void kernel_launch(void* const* d_in, const int* in_sizes, int n_in,
                              void* d_out, int out_size, void* d_ws, size_t ws_size,
                              hipStream_t stream) {
    const float* q     = (const float*)d_in[0];
    const float* k     = (const float*)d_in[1];
    const float* v     = (const float*)d_in[2];
    const float* mask  = (const float*)d_in[3];
    const float* bias  = (const float*)d_in[4];
    const float* gamma = (const float*)d_in[5];
    const float* beta  = (const float*)d_in[6];
    const float* pw    = (const float*)d_in[7];
    const float* pb    = (const float*)d_in[8];
    float* out = (float*)d_out;

    const int nw = in_sizes[0] / (NTOK * EMB);   // 4096 windows
    hipLaunchKernelGGL(winattn_fused, dim3(nw), dim3(192), 0, stream,
                       q, k, v, mask, bias, gamma, beta, pw, pb, out);
}
